// PaiNN_74698071212154
// MI455X (gfx1250) — compile-verified, matrix operand order checked by reading, not run on hardware
//
#include <hip/hip_runtime.h>

#define N_ATOMS 10000
#define N_EDGES 150000
#define NB      128
#define NRBF    20
#define CUTOFF  5.0f
#define EPSV    1e-08f
#define NBATCH  64

typedef __attribute__((ext_vector_type(16))) _Float16 v16h;
typedef __attribute__((ext_vector_type(8)))  float    v8f;

// Fast silu: single v_rcp_f32 instead of IEEE div_scale/div_fmas sequence.
__device__ __forceinline__ float silu(float x) {
    return x * __builtin_amdgcn_rcpf(1.0f + __expf(-x));
}

// ---------------------------------------------------------------------------
// Generic WMMA GEMM: C[M x N] = act(A[M x K] @ W[K x N] + bias)
// REQUIREMENTS (all call sites satisfy these): M % 16 == 0, K % 32 == 0,
// N % 64 == 0, A/W/C 16-byte aligned. No bounds guards -> unconditional
// vector loads (global_load_b128) instead of per-lane predicated dword loads.
// One wave computes one 16x16 output tile; block = 128 threads (4 waves along N).
// grid = (M/16, N/64). f32 memory -> f16 fragments -> f32 WMMA accumulate.
// ---------------------------------------------------------------------------
template<int ACT, int HASB>
__global__ void wmma_gemm_kernel(const float* __restrict__ A,
                                 const float* __restrict__ W,
                                 const float* __restrict__ bias,
                                 float* __restrict__ C,
                                 int M, int N, int K) {
    const int lane = threadIdx.x & 31;
    const int wave = threadIdx.x >> 5;
    const int lo   = lane & 15;
    const int hi   = lane >> 4;          // 0 or 1 (half-wave)
    const int row0 = blockIdx.x * 16;
    const int col0 = blockIdx.y * 64 + wave * 16;
    const int row  = row0 + lo;          // A-fragment row for this lane
    const int col  = col0 + lo;          // B/C-fragment column for this lane

    v8f acc = {};

    for (int k0 = 0; k0 < K; k0 += 32) {
        // A fragment (16x32 f16): lane<16 holds K {0..7,16..23}, lane>=16 {8..15,24..31}
        // Two contiguous 32B chunks per lane -> global_load_b128 x4.
        const float* arow = A + (size_t)row * K + k0 + hi * 8;
        const float4 pa0 = *(const float4*)(arow + 0);
        const float4 pa1 = *(const float4*)(arow + 4);
        const float4 pa2 = *(const float4*)(arow + 16);
        const float4 pa3 = *(const float4*)(arow + 20);
        v16h a;
        a[0]  = (_Float16)pa0.x; a[1]  = (_Float16)pa0.y;
        a[2]  = (_Float16)pa0.z; a[3]  = (_Float16)pa0.w;
        a[4]  = (_Float16)pa1.x; a[5]  = (_Float16)pa1.y;
        a[6]  = (_Float16)pa1.z; a[7]  = (_Float16)pa1.w;
        a[8]  = (_Float16)pa2.x; a[9]  = (_Float16)pa2.y;
        a[10] = (_Float16)pa2.z; a[11] = (_Float16)pa2.w;
        a[12] = (_Float16)pa3.x; a[13] = (_Float16)pa3.y;
        a[14] = (_Float16)pa3.z; a[15] = (_Float16)pa3.w;

        // B fragment (32x16 f16): symmetric K distribution, column = lane lo.
        // Per-element stride N, but lane-contiguous -> coalesced dword loads.
        const float* wcol = W + (size_t)(k0 + hi * 8) * N + col;
        v16h b;
        #pragma unroll
        for (int e = 0; e < 16; ++e) {
            const int kl = e + (e >= 8 ? 8 : 0);
            b[e] = (_Float16)wcol[(size_t)kl * N];
        }

        acc = __builtin_amdgcn_wmma_f32_16x16x32_f16(
            /*neg_a=*/false, a, /*neg_b=*/false, b,
            /*c_mod=*/(short)0, acc, /*reuse_a=*/false, /*reuse_b=*/false);
    }

    const float bs = HASB ? bias[col] : 0.0f;
    #pragma unroll
    for (int r = 0; r < 8; ++r) {
        const int orow = row0 + r + hi * 8;   // C layout: vgpr r holds row r (+8 hi half)
        float v = acc[r] + bs;
        if (ACT) v = silu(v);
        C[(size_t)orow * N + col] = v;
    }
}

// ---------------------------------------------------------------------------
// Per-edge geometry: r_ij, d_ij, dir, fcut, phi*fcut (so filters = phiF@fW + fb*fcut)
// ---------------------------------------------------------------------------
__global__ void edge_geom_kernel(const float* __restrict__ pos,
                                 const int* __restrict__ idx_i,
                                 const int* __restrict__ idx_j,
                                 float* __restrict__ phiF,
                                 float* __restrict__ dirv,
                                 float* __restrict__ fcv,
                                 int E) {
    int e = blockIdx.x * blockDim.x + threadIdx.x;
    if (e >= E) return;
    const int i = idx_i[e], j = idx_j[e];
    const float rx = pos[j * 3 + 0] - pos[i * 3 + 0];
    const float ry = pos[j * 3 + 1] - pos[i * 3 + 1];
    const float rz = pos[j * 3 + 2] - pos[i * 3 + 2];
    const float d   = sqrtf(rx * rx + ry * ry + rz * rz);
    const float inv = 1.0f / d;
    dirv[(size_t)e * 3 + 0] = rx * inv;
    dirv[(size_t)e * 3 + 1] = ry * inv;
    dirv[(size_t)e * 3 + 2] = rz * inv;
    const float fc = (d < CUTOFF) ? 0.5f * (__cosf(d * 3.14159265358979323846f / CUTOFF) + 1.0f)
                                  : 0.0f;
    fcv[e] = fc;
    const float w     = CUTOFF / (float)(NRBF - 1);
    const float coeff = -0.5f / (w * w);
    #pragma unroll
    for (int k = 0; k < NRBF; ++k) {
        const float t = d - (float)k * w;
        phiF[(size_t)e * NRBF + k] = fc * __expf(coeff * t * t);
    }
}

// ---------------------------------------------------------------------------
// q initialization: q[a][c] = emb[z[a]][c]
// ---------------------------------------------------------------------------
__global__ void init_q_kernel(const float* __restrict__ emb,
                              const int* __restrict__ z,
                              float* __restrict__ q, int n) {
    int idx = blockIdx.x * blockDim.x + threadIdx.x;
    if (idx >= n) return;
    const int a = idx >> 7, c = idx & 127;
    q[idx] = emb[(size_t)z[a] * NB + c];
}

// ---------------------------------------------------------------------------
// Per-edge message: regenerate 384-wide filter slice from phiF (K=20), multiply
// with gathered x[idx_j], scatter-add dq into q and dmu into mu_new (gather mu_old).
// One block (128 threads = feature channels) per edge.
// ---------------------------------------------------------------------------
__global__ void edge_message_kernel(const float* __restrict__ phiF,
                                    const float* __restrict__ fcv,
                                    const float* __restrict__ dirv,
                                    const float* __restrict__ fW,
                                    const float* __restrict__ fb,
                                    int it,
                                    const float* __restrict__ x_atom,   // [A,384]
                                    const int* __restrict__ idx_i,
                                    const int* __restrict__ idx_j,
                                    const float* __restrict__ mu_old,   // [A,3,128]
                                    float* __restrict__ mu_new,         // [A,3,128]
                                    float* __restrict__ q) {            // [A,128]
    __shared__ float sphi[NRBF];
    __shared__ float sdir[3];
    __shared__ float sfc;
    const int e = blockIdx.x;
    const int c = threadIdx.x;
    if (c < NRBF)          sphi[c] = phiF[(size_t)e * NRBF + c];
    else if (c == NRBF)    sfc     = fcv[e];
    else if (c < NRBF + 4) sdir[c - NRBF - 1] = dirv[(size_t)e * 3 + (c - NRBF - 1)];
    __syncthreads();

    const int ia = idx_i[e], ja = idx_j[e];
    const float fc = sfc;
    float f0 = fb[it * 384 + c]       * fc;
    float f1 = fb[it * 384 + 128 + c] * fc;
    float f2 = fb[it * 384 + 256 + c] * fc;
    #pragma unroll 4
    for (int k = 0; k < NRBF; ++k) {
        const float p = sphi[k];
        const float* wr = fW + (size_t)k * (3 * 3 * NB) + it * 384;
        f0 = fmaf(p, wr[c],       f0);
        f1 = fmaf(p, wr[c + 128], f1);
        f2 = fmaf(p, wr[c + 256], f2);
    }
    const float* xj = x_atom + (size_t)ja * 384;
    const float dq    = f0 * xj[c];
    const float dmuR  = f1 * xj[128 + c];
    const float dmumu = f2 * xj[256 + c];
    atomicAdd(&q[(size_t)ia * NB + c], dq);
    const float* mj = mu_old + (size_t)ja * 384;
    float*       mn = mu_new + (size_t)ia * 384;
    #pragma unroll
    for (int d = 0; d < 3; ++d)
        atomicAdd(&mn[d * NB + c], dmuR * sdir[d] + dmumu * mj[d * NB + c]);
}

// ---------------------------------------------------------------------------
// Context build: mu_Vn = ||mu_V||_3dims, svWc = sum_d mu_V*mu_Wc, ctx = [q, mu_Vn]
// mu_mix layout: [(a*3+d) * 256], cols 0..127 = mu_V, 128..255 = mu_Wc
// ---------------------------------------------------------------------------
__global__ void ctx_build_kernel(const float* __restrict__ mu_mix,
                                 const float* __restrict__ q,
                                 float* __restrict__ ctx,    // [A,256]
                                 float* __restrict__ svWc,   // [A,128]
                                 int n) {
    int idx = blockIdx.x * blockDim.x + threadIdx.x;
    if (idx >= n) return;
    const int a = idx >> 7, c = idx & 127;
    float ssq = EPSV, svw = 0.0f;
    #pragma unroll
    for (int d = 0; d < 3; ++d) {
        const float v = mu_mix[(size_t)(a * 3 + d) * 256 + c];
        const float w = mu_mix[(size_t)(a * 3 + d) * 256 + 128 + c];
        ssq = fmaf(v, v, ssq);
        svw = fmaf(v, w, svw);
    }
    svWc[idx] = svw;
    ctx[(size_t)a * 256 + c]       = q[idx];
    ctx[(size_t)a * 256 + 128 + c] = sqrtf(ssq);
}

// ---------------------------------------------------------------------------
// Interaction update: q += dq_in + dqmu_in*svWc ; mu += dmu_in*mu_Wc (broadcast d)
// xctx layout [A,384]: cols 0..127 dq_in, 128..255 dmu_in, 256..383 dqmu_in
// ---------------------------------------------------------------------------
__global__ void update_kernel(const float* __restrict__ xctx,
                              const float* __restrict__ svWc,
                              const float* __restrict__ mu_mix,
                              float* __restrict__ q,
                              float* __restrict__ mu,
                              int n) {
    int idx = blockIdx.x * blockDim.x + threadIdx.x;
    if (idx >= n) return;
    const int a = idx >> 7, c = idx & 127;
    const float dq_in   = xctx[(size_t)a * 384 + c];
    const float dmu_in  = xctx[(size_t)a * 384 + 128 + c];
    const float dqmu_in = xctx[(size_t)a * 384 + 256 + c];
    q[idx] += dq_in + dqmu_in * svWc[idx];
    #pragma unroll
    for (int d = 0; d < 3; ++d) {
        const float wc = mu_mix[(size_t)(a * 3 + d) * 256 + 128 + c];
        mu[(size_t)a * 384 + d * NB + c] += dmu_in * wc;
    }
}

// ---------------------------------------------------------------------------
// Output head tail: y_atom = h2 . o3_W + o3_b ; segment-sum into y[idx_m[a]]
// One wave per atom (lane-strided dot + shuffle reduce).
// ---------------------------------------------------------------------------
__global__ void out_reduce_kernel(const float* __restrict__ h2,
                                  const float* __restrict__ o3W,
                                  const float* __restrict__ o3b,
                                  const int* __restrict__ idx_m,
                                  float* __restrict__ y, int M) {
    const int gwave = (blockIdx.x * blockDim.x + threadIdx.x) >> 5;
    const int lane  = threadIdx.x & 31;
    if (gwave >= M) return;
    float s = 0.0f;
    #pragma unroll
    for (int k = lane; k < NB; k += 32)
        s = fmaf(h2[(size_t)gwave * NB + k], o3W[k], s);
    #pragma unroll
    for (int off = 16; off; off >>= 1)
        s += __shfl_down(s, off, 32);
    if (lane == 0)
        atomicAdd(&y[idx_m[gwave]], s + o3b[0]);
}

// ---------------------------------------------------------------------------
extern "C" void kernel_launch(void* const* d_in, const int* in_sizes, int n_in,
                              void* d_out, int out_size, void* d_ws, size_t ws_size,
                              hipStream_t stream) {
    const float* pos   = (const float*)d_in[0];
    const float* emb   = (const float*)d_in[1];
    const float* fW    = (const float*)d_in[2];
    const float* fb    = (const float*)d_in[3];
    const float* ic1W  = (const float*)d_in[4];
    const float* ic1b  = (const float*)d_in[5];
    const float* ic2W  = (const float*)d_in[6];
    const float* ic2b  = (const float*)d_in[7];
    const float* mixW  = (const float*)d_in[8];
    const float* ctx1W = (const float*)d_in[9];
    const float* ctx1b = (const float*)d_in[10];
    const float* ctx2W = (const float*)d_in[11];
    const float* ctx2b = (const float*)d_in[12];
    const float* o1W   = (const float*)d_in[13];
    const float* o1b   = (const float*)d_in[14];
    const float* o2W   = (const float*)d_in[15];
    const float* o2b   = (const float*)d_in[16];
    const float* o3W   = (const float*)d_in[17];
    const float* o3b   = (const float*)d_in[18];
    const int*   zat   = (const int*)d_in[19];
    const int*   idx_i = (const int*)d_in[20];
    const int*   idx_j = (const int*)d_in[21];
    const int*   idx_m = (const int*)d_in[22];

    // workspace carve-up (floats)
    float* p = (float*)d_ws;
    float* phiF   = p; p += (size_t)N_EDGES * NRBF;        // 3.0M
    float* dirv   = p; p += (size_t)N_EDGES * 3;           // 0.45M
    float* fcv    = p; p += (size_t)N_EDGES;               // 0.15M
    float* q      = p; p += (size_t)N_ATOMS * NB;          // 1.28M
    float* muA    = p; p += (size_t)N_ATOMS * 3 * NB;      // 3.84M
    float* muB    = p; p += (size_t)N_ATOMS * 3 * NB;      // 3.84M
    float* xbuf   = p; p += (size_t)N_ATOMS * 384;         // 3.84M
    float* h1     = p; p += (size_t)N_ATOMS * NB;          // 1.28M
    float* h2     = p; p += (size_t)N_ATOMS * NB;          // 1.28M
    float* mu_mix = p; p += (size_t)N_ATOMS * 3 * 256;     // 7.68M
    float* ctxb   = p; p += (size_t)N_ATOMS * 256;         // 2.56M
    float* svWc   = p; p += (size_t)N_ATOMS * NB;          // 1.28M

    const size_t MU_BYTES = (size_t)N_ATOMS * 3 * NB * sizeof(float);
    const int    NAC      = N_ATOMS * NB;    // 1,280,000 per-(atom,channel) items

    hipMemsetAsync(d_out, 0, NBATCH * sizeof(float), stream);
    hipMemsetAsync(muA, 0, MU_BYTES, stream);

    init_q_kernel<<<(NAC + 255) / 256, 256, 0, stream>>>(emb, zat, q, NAC);
    edge_geom_kernel<<<(N_EDGES + 255) / 256, 256, 0, stream>>>(
        pos, idx_i, idx_j, phiF, dirv, fcv, N_EDGES);

    float* mu_cur = muA;
    float* mu_nxt = muB;

    for (int it = 0; it < 3; ++it) {
        // x = silu(q @ ic1_W + b) @ ic2_W + b
        wmma_gemm_kernel<1, 1><<<dim3(N_ATOMS / 16, 2), 128, 0, stream>>>(
            q, ic1W + (size_t)it * NB * NB, ic1b + it * NB, h1, N_ATOMS, NB, NB);
        wmma_gemm_kernel<0, 1><<<dim3(N_ATOMS / 16, 6), 128, 0, stream>>>(
            h1, ic2W + (size_t)it * NB * 384, ic2b + it * 384, xbuf, N_ATOMS, 384, NB);

        // mu_new = mu_old ; edge scatter adds dq into q and dmu into mu_new
        hipMemcpyAsync(mu_nxt, mu_cur, MU_BYTES, hipMemcpyDeviceToDevice, stream);
        edge_message_kernel<<<N_EDGES, 128, 0, stream>>>(
            phiF, fcv, dirv, fW, fb, it, xbuf, idx_i, idx_j, mu_cur, mu_nxt, q);
        { float* t = mu_cur; mu_cur = mu_nxt; mu_nxt = t; }

        // mu_mix = mu @ mix_W  (30000 x 128 @ 128 x 256), no bias
        wmma_gemm_kernel<0, 0><<<dim3((N_ATOMS * 3) / 16, 4), 128, 0, stream>>>(
            mu_cur, mixW + (size_t)it * NB * 256, nullptr, mu_mix, N_ATOMS * 3, 256, NB);

        ctx_build_kernel<<<(NAC + 255) / 256, 256, 0, stream>>>(mu_mix, q, ctxb, svWc, NAC);

        // x = silu(ctx @ ctx1_W + b) @ ctx2_W + b
        wmma_gemm_kernel<1, 1><<<dim3(N_ATOMS / 16, 2), 128, 0, stream>>>(
            ctxb, ctx1W + (size_t)it * 256 * NB, ctx1b + it * NB, h1, N_ATOMS, NB, 256);
        wmma_gemm_kernel<0, 1><<<dim3(N_ATOMS / 16, 6), 128, 0, stream>>>(
            h1, ctx2W + (size_t)it * NB * 384, ctx2b + it * 384, xbuf, N_ATOMS, 384, NB);

        update_kernel<<<(NAC + 255) / 256, 256, 0, stream>>>(
            xbuf, svWc, mu_mix, q, mu_cur, NAC);
    }

    // output head: silu(silu(q@o1+b)@o2+b) @ o3 + b, segment-sum by idx_m
    wmma_gemm_kernel<1, 1><<<dim3(N_ATOMS / 16, 2), 128, 0, stream>>>(
        q, o1W, o1b, h1, N_ATOMS, NB, NB);
    wmma_gemm_kernel<1, 1><<<dim3(N_ATOMS / 16, 2), 128, 0, stream>>>(
        h1, o2W, o2b, h2, N_ATOMS, NB, NB);
    out_reduce_kernel<<<(N_ATOMS * 32 + 255) / 256, 256, 0, stream>>>(
        h2, o3W, o3b, idx_m, (float*)d_out, N_ATOMS);

    // second output: final mu tensor
    hipMemcpyAsync((float*)d_out + NBATCH, mu_cur, MU_BYTES,
                   hipMemcpyDeviceToDevice, stream);
}
